// CausalSelfAttention_34024730919655
// MI455X (gfx1250) — compile-verified, matrix operand order checked by reading, not run on hardware
//
#include <hip/hip_runtime.h>

// ---------------------------------------------------------------------------
// CDNA5 (gfx1250, wave32) causal self-attention:
//   qkv = x @ W_attn ; flash-attention per head ; out = y @ W_proj
// GEMMs on v_wmma_f32_16x16x32_bf16 (f32 accum). bf16-copy staging uses
// GLOBAL_LOAD_ASYNC_TO_LDS_B128 (ASYNCcnt); all tiles double-buffered so
// next-tile HBM traffic overlaps WMMA work. f32->bf16 via v_cvt_pk_bf16_f32.
// ---------------------------------------------------------------------------

typedef __attribute__((ext_vector_type(16))) __bf16        bf16x16;
typedef __attribute__((ext_vector_type(2)))  __bf16        bf16x2;
typedef __attribute__((ext_vector_type(8)))  float         f32x8;
typedef __attribute__((ext_vector_type(4)))  float         f32x4;
typedef __attribute__((ext_vector_type(2)))  float         f32x2;
typedef __attribute__((ext_vector_type(4)))  unsigned int  u32x4;

#define NB 4
#define NT 2048
#define NC 1024
#define NH 16
#define ND 64
#define M1 (NB * NT)   // 8192 rows
#define N1 (3 * NC)    // 3072 qkv cols

// GEMM tiling: block tile 128(M) x 64(N) x 32(K); 8 waves, wave tile 32x32.
#define BM 128
#define BN 64
#define BK 32
#define KP 40          // LDS row stride (ushorts): 80B, 16B aligned

__device__ __forceinline__ unsigned short f2bf(float f) {
  __bf16 h = (__bf16)f;                       // native v_cvt on gfx1250
  return __builtin_bit_cast(unsigned short, h);
}
__device__ __forceinline__ unsigned pack2(float lo, float hi) {
  f32x2 v; v.x = lo; v.y = hi;
  bf16x2 h = __builtin_convertvector(v, bf16x2);   // v_cvt_pk_bf16_f32
  return __builtin_bit_cast(unsigned, h);
}

__device__ __forceinline__ bf16x16 make_frag(u32x4 lo, u32x4 hi) {
  union { u32x4 q[2]; bf16x16 v; } un;
  un.q[0] = lo; un.q[1] = hi;
  return un.v;
}

// Async 16B global->LDS copy (GVS mode: sgpr base + vgpr byte offset).
__device__ __forceinline__ void async_copy_b128(unsigned ldsAddr,
                                                const void* sbase,
                                                unsigned voff) {
  asm volatile("global_load_async_to_lds_b128 %0, %1, %2"
               :: "v"(ldsAddr), "v"(voff), "s"(sbase) : "memory");
}
__device__ __forceinline__ void wait_async0() {
  asm volatile("s_wait_asynccnt 0x0" ::: "memory");
}
__device__ __forceinline__ void wait_async4() {   // async loads retire in order
  asm volatile("s_wait_asynccnt 0x4" ::: "memory");
}
__device__ __forceinline__ unsigned lds_off(const void* p) {
  return (unsigned)(unsigned long long)p;   // low 32 bits = LDS byte offset
}

#define WMMA_BF16(A, B, C) \
  __builtin_amdgcn_wmma_f32_16x16x32_bf16(false, (A), false, (B), (short)0, (C), false, false)

// ---------------------------------------------------------------------------
// Kernel 1: qkv = x(f32) @ W_attn(f32), scatter bf16 Q/K [B,H,T,D], Vt [B,H,D,T]
// Double-buffered LDS; next-tile global loads issued before the WMMA block.
// ---------------------------------------------------------------------------
__global__ __launch_bounds__(256)
void qkv_gemm_kernel(const float* __restrict__ x, const float* __restrict__ Wa,
                     unsigned short* __restrict__ Qb, unsigned short* __restrict__ Kb,
                     unsigned short* __restrict__ Vtb) {
  __shared__ unsigned short As[2][BM][KP];   // [m][k] bf16
  __shared__ unsigned short Bs[2][BN][KP];   // [n][k] bf16

  const int tid  = threadIdx.x;
  const int lane = tid & 31, wv = tid >> 5;
  const int wm = wv & 3, wn = wv >> 2;        // wave grid 4(M) x 2(N)
  const int half = lane >> 4, l15 = lane & 15;
  const int mBase = blockIdx.y * BM;
  const int nBase = blockIdx.x * BN;

  f32x8 c00 = {}, c01 = {}, c10 = {}, c11 = {};

  auto loadA = [&](int k0, f32x4 av[4]) {
#pragma unroll
    for (int i = 0; i < 4; ++i) {
      int idx = tid + i * 256, row = idx >> 3, kq = idx & 7;
      av[i] = *(const f32x4*)(x + (size_t)(mBase + row) * NC + k0 + kq * 4);
    }
  };
  auto storeA = [&](const f32x4 av[4], int buf) {
#pragma unroll
    for (int i = 0; i < 4; ++i) {
      int idx = tid + i * 256, row = idx >> 3, kq = idx & 7;
      unsigned long long pk =
          (unsigned long long)pack2(av[i].x, av[i].y) |
          ((unsigned long long)pack2(av[i].z, av[i].w) << 32);
      *(unsigned long long*)&As[buf][row][kq * 4] = pk;   // ds_store_b64
    }
  };
  auto loadB = [&](int k0, f32x4 bv[2]) {     // vectorized along n
#pragma unroll
    for (int i = 0; i < 2; ++i) {
      int idx = tid + i * 256, n4 = idx & 15, kk = idx >> 4;
      bv[i] = *(const f32x4*)(Wa + (size_t)(k0 + kk) * N1 + nBase + n4 * 4);
    }
  };
  auto storeB = [&](const f32x4 bv[2], int buf) {   // [n][k] transpose in LDS
#pragma unroll
    for (int i = 0; i < 2; ++i) {
      int idx = tid + i * 256, n4 = idx & 15, kk = idx >> 4;
      Bs[buf][n4 * 4 + 0][kk] = f2bf(bv[i].x);
      Bs[buf][n4 * 4 + 1][kk] = f2bf(bv[i].y);
      Bs[buf][n4 * 4 + 2][kk] = f2bf(bv[i].z);
      Bs[buf][n4 * 4 + 3][kk] = f2bf(bv[i].w);
    }
  };

  { f32x4 av[4]; f32x4 bv[2];
    loadA(0, av); loadB(0, bv); storeA(av, 0); storeB(bv, 0); }
  __syncthreads();

  int cur = 0;
  for (int k0 = 0; k0 < NC; k0 += BK) {
    const bool hasNext = (k0 + BK) < NC;        // scalar branch (k0 uniform)
    f32x4 av[4]; f32x4 bv[2];
    if (hasNext) { loadA(k0 + BK, av); loadB(k0 + BK, bv); }   // in flight

    const int ra0 = wm * 32 + l15, ra1 = ra0 + 16;
    const int rb0 = wn * 32 + l15, rb1 = rb0 + 16;
    bf16x16 a0 = make_frag(*(const u32x4*)&As[cur][ra0][half * 8],
                           *(const u32x4*)&As[cur][ra0][16 + half * 8]);
    bf16x16 a1 = make_frag(*(const u32x4*)&As[cur][ra1][half * 8],
                           *(const u32x4*)&As[cur][ra1][16 + half * 8]);
    bf16x16 b0 = make_frag(*(const u32x4*)&Bs[cur][rb0][half * 8],
                           *(const u32x4*)&Bs[cur][rb0][16 + half * 8]);
    bf16x16 b1 = make_frag(*(const u32x4*)&Bs[cur][rb1][half * 8],
                           *(const u32x4*)&Bs[cur][rb1][16 + half * 8]);
    c00 = WMMA_BF16(a0, b0, c00);
    c01 = WMMA_BF16(a0, b1, c01);
    c10 = WMMA_BF16(a1, b0, c10);
    c11 = WMMA_BF16(a1, b1, c11);

    if (hasNext) { storeA(av, cur ^ 1); storeB(bv, cur ^ 1); }
    __syncthreads();
    cur ^= 1;
  }

  // --- epilogue: readfirstlane forces tile coords scalar -> scalar branches
  auto scatter = [&](const f32x8& c, int subM, int subN) {
    const int ntile = __builtin_amdgcn_readfirstlane(nBase + wn * 32 + subN * 16);
    const int mtile = __builtin_amdgcn_readfirstlane(mBase + wm * 32 + subM * 16);
    const int which = ntile >> 10;                   // 0=q 1=k 2=v
    const int hh    = (ntile >> 6) & 15;
    const int bhh   = (mtile >> 11) * NH + hh;
    const int t0    = (mtile & 2047) + 8 * half;     // per-lane-half row base
    const int dd    = (ntile & 63) + l15;            // per-lane col
    if (which == 2) {
      // V transposed: 8 consecutive t per lane -> single b128 store
      union { u32x4 q; unsigned u[4]; } pk;
#pragma unroll
      for (int j = 0; j < 4; ++j) pk.u[j] = pack2(c[2 * j], c[2 * j + 1]);
      *(u32x4*)(Vtb + ((size_t)bhh * ND + dd) * NT + t0) = pk.q;
    } else {
      unsigned short* base =
          (which == 0 ? Qb : Kb) + ((size_t)bhh * NT + t0) * ND + dd;
#pragma unroll
      for (int r = 0; r < 8; ++r) base[(size_t)r * ND] = f2bf(c[r]);
    }
  };
  scatter(c00, 0, 0); scatter(c01, 0, 1);
  scatter(c10, 1, 0); scatter(c11, 1, 1);
}

// ---------------------------------------------------------------------------
// Kernel 2: flash attention. Block = (b,h, 64-row q tile), 4 waves; K/V tiles
// double-buffered, staged with GLOBAL_LOAD_ASYNC_TO_LDS_B128; next tile's
// asyncs issued before computing the current one (in-order retire => wait<=4).
// ---------------------------------------------------------------------------
#define DP 72   // Qs/Ks row stride (ushorts): 144B, 16B aligned
#define TP 40   // Vst/Ps row stride

__global__ __launch_bounds__(128)
void flash_attn_kernel(const unsigned short* __restrict__ Qb,
                       const unsigned short* __restrict__ Kb,
                       const unsigned short* __restrict__ Vtb,
                       unsigned short* __restrict__ Yb) {
  __shared__ unsigned short Qs[64][DP];          // [qrow][d]
  __shared__ unsigned short Ks[2][32][DP];       // [key][d]
  __shared__ unsigned short Vst[2][64][TP];      // [d][key] (V pre-transposed)
  __shared__ unsigned short Ps[4][16][TP];       // per-wave P restage

  const int tid  = threadIdx.x;
  const int lane = tid & 31, wv = tid >> 5;
  const int half = lane >> 4, l15 = lane & 15;
  const int qt    = blockIdx.x & 31;             // T/64 = 32 q tiles
  const int bh    = blockIdx.x >> 5;             // b*NH + h
  const int qbase = qt * 64;

  auto stageKV = [&](int kb, int buf) {          // 4 async b128 per thread
    const unsigned short* kptr = Kb + ((size_t)bh * NT + kb) * ND;
#pragma unroll
    for (int i = 0; i < 2; ++i) {
      int u = tid + i * 128, row = u >> 3, dq = u & 7;
      async_copy_b128(lds_off(&Ks[buf][row][dq * 8]), kptr,
                      (unsigned)((row * ND + dq * 8) * 2));
    }
    const unsigned short* vptr = Vtb + (size_t)bh * ND * NT + kb;
#pragma unroll
    for (int i = 0; i < 2; ++i) {
      int u = tid + i * 128, row = u >> 2, kq = u & 3;
      async_copy_b128(lds_off(&Vst[buf][row][kq * 8]), vptr,
                      (unsigned)((row * NT + kq * 8) * 2));
    }
  };

  // --- prologue: async-stage Q tile (4 ops) + K/V tile 0 (4 ops)
  const unsigned short* qptr = Qb + ((size_t)bh * NT + qbase) * ND;
#pragma unroll
  for (int i = 0; i < 4; ++i) {
    int u = tid + i * 128, row = u >> 3, dq = u & 7;
    async_copy_b128(lds_off(&Qs[row][dq * 8]), qptr,
                    (unsigned)((row * ND + dq * 8) * 2));
  }
  stageKV(0, 0);

  f32x8 o0 = {}, o1 = {}, o2 = {}, o3 = {};      // 16x64 output accum
  float mrow[8], lrow[8];
#pragma unroll
  for (int r = 0; r < 8; ++r) { mrow[r] = -1e30f; lrow[r] = 0.f; }

  const int qr0 = wv * 16;                       // this wave's q stripe
  const int nkt = (qbase >> 5) + 2;              // # of 32-key tiles (>= 2)

  int cur = 0;
  for (int kt = 0; kt < nkt; ++kt) {
    const int kb = kt * 32;
    if (kt + 1 < nkt) { stageKV(kb + 32, cur ^ 1); wait_async4(); }
    else              { wait_async0(); }
    __syncthreads();                             // all waves' tile-kt asyncs done

    // --- S = Q K^T: two 16x16 key sub-tiles, D=64 = 2 WMMA K-chunks
    const int ra = qr0 + l15;
    bf16x16 aLo = make_frag(*(const u32x4*)&Qs[ra][half * 8],
                            *(const u32x4*)&Qs[ra][16 + half * 8]);
    bf16x16 aHi = make_frag(*(const u32x4*)&Qs[ra][32 + half * 8],
                            *(const u32x4*)&Qs[ra][48 + half * 8]);
    f32x8 s0 = {}, s1 = {};
    {
      int rk = l15;
      bf16x16 bLo = make_frag(*(const u32x4*)&Ks[cur][rk][half * 8],
                              *(const u32x4*)&Ks[cur][rk][16 + half * 8]);
      bf16x16 bHi = make_frag(*(const u32x4*)&Ks[cur][rk][32 + half * 8],
                              *(const u32x4*)&Ks[cur][rk][48 + half * 8]);
      s0 = WMMA_BF16(aLo, bLo, s0);
      s0 = WMMA_BF16(aHi, bHi, s0);
    }
    {
      int rk = 16 + l15;
      bf16x16 bLo = make_frag(*(const u32x4*)&Ks[cur][rk][half * 8],
                              *(const u32x4*)&Ks[cur][rk][16 + half * 8]);
      bf16x16 bHi = make_frag(*(const u32x4*)&Ks[cur][rk][32 + half * 8],
                              *(const u32x4*)&Ks[cur][rk][48 + half * 8]);
      s1 = WMMA_BF16(aLo, bLo, s1);
      s1 = WMMA_BF16(aHi, bHi, s1);
    }

    // --- causal mask + online softmax (row groups = 16 lanes per half)
    const int col0 = kb + l15, col1 = kb + 16 + l15;
    float p0[8], p1[8], alpha[8];
#pragma unroll
    for (int r = 0; r < 8; ++r) {
      int qrow = qbase + qr0 + r + 8 * half;
      float v0 = (col0 <= qrow) ? s0[r] * 0.125f : -1e30f;   // 1/sqrt(64)
      float v1 = (col1 <= qrow) ? s1[r] * 0.125f : -1e30f;
      float mx = fmaxf(v0, v1);
      mx = fmaxf(mx, __shfl_xor(mx, 1, 32));
      mx = fmaxf(mx, __shfl_xor(mx, 2, 32));
      mx = fmaxf(mx, __shfl_xor(mx, 4, 32));
      mx = fmaxf(mx, __shfl_xor(mx, 8, 32));
      float nm = fmaxf(mrow[r], mx);
      float a  = __expf(mrow[r] - nm);
      float e0 = __expf(v0 - nm);
      float e1 = __expf(v1 - nm);
      float rs = e0 + e1;
      rs += __shfl_xor(rs, 1, 32);
      rs += __shfl_xor(rs, 2, 32);
      rs += __shfl_xor(rs, 4, 32);
      rs += __shfl_xor(rs, 8, 32);
      mrow[r]  = nm;
      lrow[r]  = lrow[r] * a + rs;
      alpha[r] = a;
      p0[r] = e0; p1[r] = e1;
    }
#pragma unroll
    for (int r = 0; r < 8; ++r) {
      o0[r] *= alpha[r]; o1[r] *= alpha[r];
      o2[r] *= alpha[r]; o3[r] *= alpha[r];
    }

    // --- restage P through LDS: C-layout -> A-layout
#pragma unroll
    for (int r = 0; r < 8; ++r) {
      int row = r + 8 * half;
      Ps[wv][row][l15]      = f2bf(p0[r]);
      Ps[wv][row][16 + l15] = f2bf(p1[r]);
    }
    asm volatile("s_wait_dscnt 0x0" ::: "memory");  // wave-private RAW via LDS
    bf16x16 ap = make_frag(*(const u32x4*)&Ps[wv][l15][half * 8],
                           *(const u32x4*)&Ps[wv][l15][16 + half * 8]);

    // --- O += P @ V : 4 d-tiles of 16 cols
#pragma unroll
    for (int dt = 0; dt < 4; ++dt) {
      int rd = dt * 16 + l15;
      bf16x16 bv = make_frag(*(const u32x4*)&Vst[cur][rd][half * 8],
                             *(const u32x4*)&Vst[cur][rd][16 + half * 8]);
      f32x8 acc = (dt == 0) ? o0 : (dt == 1) ? o1 : (dt == 2) ? o2 : o3;
      acc = WMMA_BF16(ap, bv, acc);
      if (dt == 0) o0 = acc; else if (dt == 1) o1 = acc;
      else if (dt == 2) o2 = acc; else o3 = acc;
    }
    __syncthreads();                // buffer cur free for async reuse
    cur ^= 1;
  }

  // --- normalize and store y (bf16, [B,T,C] with C = h*64 + d)
  const int b = bh >> 4, h = bh & 15;
#pragma unroll
  for (int r = 0; r < 8; ++r) {
    int tt = qbase + qr0 + r + 8 * half;
    float inv = 1.0f / lrow[r];
    size_t base = ((size_t)b * NT + tt) * NC + (size_t)h * ND;
    Yb[base +  0 + l15] = f2bf(o0[r] * inv);
    Yb[base + 16 + l15] = f2bf(o1[r] * inv);
    Yb[base + 32 + l15] = f2bf(o2[r] * inv);
    Yb[base + 48 + l15] = f2bf(o3[r] * inv);
  }
}

// ---------------------------------------------------------------------------
// Kernel 3: out(f32) = y(bf16) @ W_proj(f32); A tile async-to-LDS,
// double-buffered, wait pushed after the WMMA block.
// ---------------------------------------------------------------------------
__global__ __launch_bounds__(256)
void proj_gemm_kernel(const unsigned short* __restrict__ Yb,
                      const float* __restrict__ Wp, float* __restrict__ out) {
  __shared__ unsigned short As[2][BM][KP];
  __shared__ unsigned short Bs[2][BN][KP];

  const int tid  = threadIdx.x;
  const int lane = tid & 31, wv = tid >> 5;
  const int wm = wv & 3, wn = wv >> 2;
  const int half = lane >> 4, l15 = lane & 15;
  const int mBase = blockIdx.y * BM;
  const int nBase = blockIdx.x * BN;

  f32x8 c00 = {}, c01 = {}, c10 = {}, c11 = {};

  auto asyncA = [&](int k0, int buf) {
#pragma unroll
    for (int i = 0; i < 2; ++i) {
      int idx = tid + i * 256, row = idx >> 2, kq = idx & 3;
      async_copy_b128(lds_off(&As[buf][row][kq * 8]), Yb,
                      (unsigned)(((size_t)(mBase + row) * NC + k0 + kq * 8) * 2));
    }
  };
  auto loadB = [&](int k0, f32x4 bv[2]) {
#pragma unroll
    for (int i = 0; i < 2; ++i) {
      int idx = tid + i * 256, n4 = idx & 15, kk = idx >> 4;
      bv[i] = *(const f32x4*)(Wp + (size_t)(k0 + kk) * NC + nBase + n4 * 4);
    }
  };
  auto storeB = [&](const f32x4 bv[2], int buf) {
#pragma unroll
    for (int i = 0; i < 2; ++i) {
      int idx = tid + i * 256, n4 = idx & 15, kk = idx >> 4;
      Bs[buf][n4 * 4 + 0][kk] = f2bf(bv[i].x);
      Bs[buf][n4 * 4 + 1][kk] = f2bf(bv[i].y);
      Bs[buf][n4 * 4 + 2][kk] = f2bf(bv[i].z);
      Bs[buf][n4 * 4 + 3][kk] = f2bf(bv[i].w);
    }
  };

  { f32x4 bv[2];
    asyncA(0, 0); loadB(0, bv); storeB(bv, 0); wait_async0(); }
  __syncthreads();

  int cur = 0;
  for (int k0 = 0; k0 < NC; k0 += BK) {
    const bool hasNext = (k0 + BK) < NC;
    f32x4 bv[2];
    if (hasNext) { asyncA(k0 + BK, cur ^ 1); loadB(k0 + BK, bv); }

    const int ra0 = wm * 32 + l15, ra1 = ra0 + 16;
    const int rb0 = wn * 32 + l15, rb1 = rb0 + 16;
    bf16x16 a0 = make_frag(*(const u32x4*)&As[cur][ra0][half * 8],
                           *(const u32x4*)&As[cur][ra0][16 + half * 8]);
    bf16x16 a1 = make_frag(*(const u32x4*)&As[cur][ra1][half * 8],
                           *(const u32x4*)&As[cur][ra1][16 + half * 8]);
    bf16x16 b0 = make_frag(*(const u32x4*)&Bs[cur][rb0][half * 8],
                           *(const u32x4*)&Bs[cur][rb0][16 + half * 8]);
    bf16x16 b1 = make_frag(*(const u32x4*)&Bs[cur][rb1][half * 8],
                           *(const u32x4*)&Bs[cur][rb1][16 + half * 8]);
    c00 = WMMA_BF16(a0, b0, c00);
    c01 = WMMA_BF16(a0, b1, c01);
    c10 = WMMA_BF16(a1, b0, c10);
    c11 = WMMA_BF16(a1, b1, c11);

    if (hasNext) { storeB(bv, cur ^ 1); wait_async0(); }
    __syncthreads();
    cur ^= 1;
  }

  auto store = [&](const f32x8& c, int subM, int subN) {
    int n = nBase + wn * 32 + subN * 16 + l15;
#pragma unroll
    for (int r = 0; r < 8; ++r) {
      int m = mBase + wm * 32 + subM * 16 + r + 8 * half;
      out[(size_t)m * NC + n] = c[r];
    }
  };
  store(c00, 0, 0); store(c01, 0, 1);
  store(c10, 1, 0); store(c11, 1, 1);
}

// ---------------------------------------------------------------------------
extern "C" void kernel_launch(void* const* d_in, const int* in_sizes, int n_in,
                              void* d_out, int out_size, void* d_ws, size_t ws_size,
                              hipStream_t stream) {
  (void)in_sizes; (void)n_in; (void)out_size; (void)ws_size;
  const float* x  = (const float*)d_in[0];
  const float* Wa = (const float*)d_in[1];
  const float* Wp = (const float*)d_in[2];
  float* out = (float*)d_out;

  char* ws = (char*)d_ws;
  const size_t sz = (size_t)NB * NH * NT * ND * sizeof(unsigned short); // 16 MiB
  unsigned short* Qb  = (unsigned short*)(ws);
  unsigned short* Kb  = (unsigned short*)(ws + sz);
  unsigned short* Vtb = (unsigned short*)(ws + 2 * sz);
  unsigned short* Yb  = (unsigned short*)(ws + 3 * sz);

  qkv_gemm_kernel<<<dim3(N1 / BN, M1 / BM), 256, 0, stream>>>(x, Wa, Qb, Kb, Vtb);
  flash_attn_kernel<<<dim3(NB * NH * (NT / 64)), 128, 0, stream>>>(Qb, Kb, Vtb, Yb);
  proj_gemm_kernel<<<dim3(NC / BN, M1 / BM), 256, 0, stream>>>(Yb, Wp, out);
}
